// MultiHeadedAttention_78932908966040
// MI455X (gfx1250) — matrix operand order repaired and checked by reading, not compiled, not run on hardware
//
#include <hip/hip_runtime.h>

#define B_  8
#define S_  2048
#define D_  512
#define H_  8
#define DK_ 64

typedef _Float16 h16;
typedef __attribute__((ext_vector_type(16))) _Float16 v16h;
typedef __attribute__((ext_vector_type(8)))  float    v8f;
typedef __attribute__((ext_vector_type(4)))  unsigned int v4u;
typedef __attribute__((ext_vector_type(8)))  int      v8i;
typedef __attribute__((ext_vector_type(4)))  int      v4i;

// ---- WMMA fragment loaders (wave32; contiguous k = 16*(lane/16) + j, same order in A and B) ----

// A (16x32 f16): lane 0-15 -> M=lane, K 0..15; lane 16-31 -> M=lane-16, K 16..31 (contiguous).
__device__ __forceinline__ v16h ld_a(const h16* t, int ld, int row, int kb) {
  int lane = threadIdx.x & 31;
  int b8 = (lane < 16) ? 0 : 16;
  const h16* p = t + row * ld + kb;
  v16h a;
#pragma unroll
  for (int j = 0; j < 16; ++j) {
    int k = b8 + j;
    a[j] = p[k];
  }
  return a;
}

// B (32x16 f16), tile stored as t[n][k] (k contiguous)
__device__ __forceinline__ v16h ld_b_nk(const h16* t, int ld, int n, int kb) {
  int lane = threadIdx.x & 31;
  int kbase = (lane < 16) ? 0 : 16;
  const h16* p = t + n * ld + kb + kbase;
  v16h b;
#pragma unroll
  for (int j = 0; j < 16; ++j) b[j] = p[j];
  return b;
}

// B (32x16 f16), tile stored as t[k][n] (strided reads)
__device__ __forceinline__ v16h ld_b_kn(const h16* t, int ld, int n, int kb) {
  int lane = threadIdx.x & 31;
  int kbase = (lane < 16) ? 0 : 16;
  const h16* p = t + (kb + kbase) * ld + n;
  v16h b;
#pragma unroll
  for (int j = 0; j < 16; ++j) b[j] = p[j * ld];
  return b;
}

// ---- Tensor Data Mover: 1-D f16 run, global -> LDS, with LDS padding ----
// pad_enable=1, pad_interval=4 (after every 32 DWORDs = 64 halves),
// pad_amount=3 (4 DWORDs = 8 halves)  => LDS row stride 72 halves.
__device__ __forceinline__ void tdm_load_f16(unsigned int lds_byte_off,
                                             const h16* gsrc, unsigned int nhalf) {
  unsigned long long ga = (unsigned long long)(size_t)gsrc;
  v4u g0;
  g0[0] = 1u;                                            // count=1 (valid user D#)
  g0[1] = lds_byte_off;                                  // lds_addr
  g0[2] = (unsigned int)ga;                              // global_addr[31:0]
  g0[3] = (unsigned int)((ga >> 32) & 0x01ffffffu)       // global_addr[56:32]
          | (2u << 30);                                  // type = 2 ("image")
  v8i g1;
  g1[0] = (int)((1u << 16)            // data_size = 1 (2 bytes)
              | (1u << 20)            // pad_enable
              | (4u << 22)            // pad_interval: 32 DWORDs
              | (3u << 25));          // pad_amount: 4 DWORDs
  g1[1] = (int)((nhalf & 0xffffu) << 16);                // tensor_dim0[15:0]
  g1[2] = (int)(((nhalf >> 16) & 0xffffu) | (1u << 16)); // tensor_dim0[31:16] | tensor_dim1=1
  g1[3] = (int)((nhalf & 0xffffu) << 16);                // tile_dim0 = nhalf (1-D tile)
  g1[4] = 0;                                             // tile_dim1 = tile_dim2 = 0
  g1[5] = (int)nhalf;                                    // tensor_dim0_stride[31:0]
  g1[6] = 0;
  g1[7] = 0;
  v4i z4 = {0, 0, 0, 0};
#if defined(__clang_major__) && (__clang_major__ >= 23)
  v8i z8 = {0, 0, 0, 0, 0, 0, 0, 0};
  __builtin_amdgcn_tensor_load_to_lds(g0, g1, z4, z4, z8, 0);
#else
  __builtin_amdgcn_tensor_load_to_lds(g0, g1, z4, z4, 0);
#endif
}

// ---------------- Kernel 1: per-head Q/K/V projections ----------------
__global__ __launch_bounds__(128)
void proj_kernel(const float* __restrict__ q, const float* __restrict__ k,
                 const float* __restrict__ v,
                 const float* __restrict__ Wq, const float* __restrict__ bq,
                 const float* __restrict__ Wk, const float* __restrict__ bk,
                 const float* __restrict__ Wv, const float* __restrict__ bv,
                 h16* __restrict__ qp, h16* __restrict__ kp, h16* __restrict__ vp)
{
  __shared__ __align__(16) h16 Wl[DK_ * 72];
  __shared__ __align__(16) h16 Xl[64 * 72];

  const int p = blockIdx.z;
  const float* x    = (p == 0) ? q  : (p == 1) ? k  : v;
  const float* W    = (p == 0) ? Wq : (p == 1) ? Wk : Wv;
  const float* bias = (p == 0) ? bq : (p == 1) ? bk : bv;
  h16*         out  = (p == 0) ? qp : (p == 1) ? kp : vp;

  const int bh = blockIdx.y, b = bh / H_, h = bh % H_;
  const int s0 = blockIdx.x * 64;
  const int tid = threadIdx.x;

  for (int i = tid; i < DK_ * DK_; i += 128) {
    int e = i >> 6, d = i & 63;
    Wl[e * 72 + d] = (h16)W[((size_t)h * DK_ + e) * DK_ + d];
  }
  for (int i = tid; i < 64 * DK_; i += 128) {
    int r = i >> 6, d = i & 63;
    Xl[r * 72 + d] = (h16)x[((size_t)b * S_ + (s0 + r)) * D_ + h * DK_ + d];
  }
  __syncthreads();

  const int wave = tid >> 5, lane = tid & 31;
  const int arow = wave * 16 + (lane & 15);
  v16h a0 = ld_a(Xl, 72, arow, 0);
  v16h a1 = ld_a(Xl, 72, arow, 32);

  v8f acc[4] = {};
#pragma unroll
  for (int nt = 0; nt < 4; ++nt) {
    int n = nt * 16 + (lane & 15);
    v16h b0 = ld_b_nk(Wl, 72, n, 0);   // B[k=d][n=e] = W[e][d] = Wl[n][k]
    v16h b1 = ld_b_nk(Wl, 72, n, 32);
    acc[nt] = __builtin_amdgcn_wmma_f32_16x16x32_f16(false, a0, false, b0,
                                                     (short)0, acc[nt], false, false);
    acc[nt] = __builtin_amdgcn_wmma_f32_16x16x32_f16(false, a1, false, b1,
                                                     (short)0, acc[nt], false, false);
  }

  const size_t obase = (size_t)bh * S_;
#pragma unroll
  for (int nt = 0; nt < 4; ++nt) {
    int col = nt * 16 + (lane & 15);
    float bb = bias[h * DK_ + col];
#pragma unroll
    for (int r = 0; r < 8; ++r) {
      int row = s0 + wave * 16 + r + ((lane < 16) ? 0 : 8);
      out[(obase + row) * DK_ + col] = (h16)(acc[nt][r] + bb);
    }
  }
}

// ---------------- Kernel 2: flash attention with TDM double-buffering ----------------
// grid = (S/64, B*H); block = 128 (4 waves); each wave TDM-loads its own 8-row
// slice of the NEXT 32-key K/V tile while computing on the current one.
__global__ __launch_bounds__(128)
void attn_kernel(const h16* __restrict__ qp, const h16* __restrict__ kp,
                 const h16* __restrict__ vp, h16* __restrict__ oh)
{
  __shared__ __align__(16) h16 Ql[64 * 72];
  __shared__ __align__(16) h16 Kl[2][32 * 72];
  __shared__ __align__(16) h16 Vl[2][32 * 72];
  __shared__ __align__(16) h16 Pl[4 * 16 * 40];

  const int bh = blockIdx.y, b = bh / H_, h = bh % H_;
  const int s0 = blockIdx.x * 64;
  const int tid = threadIdx.x, wave = tid >> 5, lane = tid & 31;

  const h16* Qg = qp + (size_t)bh * S_ * DK_;
  const h16* Kg = kp + (size_t)bh * S_ * DK_;
  const h16* Vg = vp + (size_t)bh * S_ * DK_;

  // per-wave slice: 8 rows x 64 halves = 512 halves = 1KB; LDS slice at stride-72 rows
  const unsigned int sl_lds = (unsigned int)(wave * 8 * 72 * 2);
  const size_t       sl_g   = (size_t)(wave * 8) * DK_;

  // prologue: TDM tile 0 into buffer 0 (each wave its slice)
  tdm_load_f16((unsigned int)(size_t)&Kl[0][0] + sl_lds, Kg + sl_g, 512);
  tdm_load_f16((unsigned int)(size_t)&Vl[0][0] + sl_lds, Vg + sl_g, 512);

  // stage Q tile (64x64 halves) with B128 transfers
  {
    const uint4* src = (const uint4*)(Qg + (size_t)s0 * DK_);
    for (int i = tid; i < 512; i += 128) {
      int r = i >> 3, c = i & 7;
      *(uint4*)&Ql[r * 72 + c * 8] = src[r * 8 + c];
    }
  }
  __syncthreads();

  const int arow = wave * 16 + (lane & 15);
  v16h qa0 = ld_a(Ql, 72, arow, 0);
  v16h qa1 = ld_a(Ql, 72, arow, 32);

  v8f o[4] = {};
  float m[8], l[8];
#pragma unroll
  for (int r = 0; r < 8; ++r) { m[r] = -1e30f; l[r] = 0.f; }

  h16* Pw = &Pl[wave * 16 * 40];

  for (int it = 0; it < S_ / 32; ++it) {
    const int buf = it & 1;
    if (it + 1 < S_ / 32) {
      // issue next tile's DMA, then wait for exactly the current tile (<=2 in flight)
      const size_t g_next = (size_t)(it + 1) * 32 * DK_ + sl_g;
      tdm_load_f16((unsigned int)(size_t)&Kl[buf ^ 1][0] + sl_lds, Kg + g_next, 512);
      tdm_load_f16((unsigned int)(size_t)&Vl[buf ^ 1][0] + sl_lds, Vg + g_next, 512);
      __builtin_amdgcn_s_wait_tensorcnt(2);
    } else {
      __builtin_amdgcn_s_wait_tensorcnt(0);
    }
    __syncthreads();   // publish current tile to all waves

    const h16* Kc = &Kl[buf][0];
    const h16* Vc = &Vl[buf][0];

    // scores: S = Q K^T   (B[k=d][n=key] = Kc[key][d])
    v8f sc[2] = {};
#pragma unroll
    for (int n2 = 0; n2 < 2; ++n2) {
      int n = n2 * 16 + (lane & 15);
      v16h kb0 = ld_b_nk(Kc, 72, n, 0);
      v16h kb1 = ld_b_nk(Kc, 72, n, 32);
      sc[n2] = __builtin_amdgcn_wmma_f32_16x16x32_f16(false, qa0, false, kb0,
                                                      (short)0, sc[n2], false, false);
      sc[n2] = __builtin_amdgcn_wmma_f32_16x16x32_f16(false, qa1, false, kb1,
                                                      (short)0, sc[n2], false, false);
    }
#pragma unroll
    for (int n2 = 0; n2 < 2; ++n2)
#pragma unroll
      for (int r = 0; r < 8; ++r) sc[n2][r] *= 0.125f;   // 1/sqrt(DK)

    // online softmax; row = r (+8 for upper half-wave); cols across 16 lanes
#pragma unroll
    for (int r = 0; r < 8; ++r) {
      float rm = fmaxf(sc[0][r], sc[1][r]);
#pragma unroll
      for (int off = 8; off > 0; off >>= 1) rm = fmaxf(rm, __shfl_xor(rm, off, 16));
      float mnew  = fmaxf(m[r], rm);
      float scale = __expf(m[r] - mnew);
      float p0 = __expf(sc[0][r] - mnew);
      float p1 = __expf(sc[1][r] - mnew);
      float rs = p0 + p1;
#pragma unroll
      for (int off = 8; off > 0; off >>= 1) rs += __shfl_xor(rs, off, 16);
      l[r] = l[r] * scale + rs;
      m[r] = mnew;
#pragma unroll
      for (int nt = 0; nt < 4; ++nt) o[nt][r] *= scale;
      int prow = r + ((lane < 16) ? 0 : 8);
      Pw[prow * 40 + (lane & 15)]      = (h16)p0;
      Pw[prow * 40 + 16 + (lane & 15)] = (h16)p1;
    }
    // intra-wave DS RAW: P stores above feed A-fragment loads below
    asm volatile("s_wait_dscnt 0" ::: "memory");

    // O += P V   (A = P 16x32, B[k=sk][n=d] = Vc[sk][d])
    v16h pa = ld_a(Pw, 40, (lane & 15), 0);
#pragma unroll
    for (int nt = 0; nt < 4; ++nt) {
      v16h vb = ld_b_kn(Vc, 72, nt * 16 + (lane & 15), 0);
      o[nt] = __builtin_amdgcn_wmma_f32_16x16x32_f16(false, pa, false, vb,
                                                     (short)0, o[nt], false, false);
    }
    __syncthreads();   // all waves done reading before next TDM overwrites buf^1
  }

  // normalize and write concatenated heads as f16: oh[b][s][h*DK + d]
#pragma unroll
  for (int nt = 0; nt < 4; ++nt) {
    int col = nt * 16 + (lane & 15);
#pragma unroll
    for (int r = 0; r < 8; ++r) {
      int row = s0 + wave * 16 + r + ((lane < 16) ? 0 : 8);
      float val = o[nt][r] / l[r];
      oh[((size_t)b * S_ + row) * D_ + h * DK_ + col] = (h16)val;
    }
  }
}

// ---------------- Kernel 3: output projection out = Oh*Wo^T + bo ----------------
__global__ __launch_bounds__(128)
void outproj_kernel(const h16* __restrict__ oh, const float* __restrict__ Wo,
                    const float* __restrict__ bo, float* __restrict__ out)
{
  __shared__ __align__(16) h16 Al[64 * 40];
  __shared__ __align__(16) h16 Bl[64 * 40];   // Bl[n][k] = Wo[n0+n][kb+k]

  const int tid = threadIdx.x, wave = tid >> 5, lane = tid & 31;
  const int row0 = blockIdx.x * 64;
  const int n0   = blockIdx.y * 64;

  v8f acc[4] = {};
  for (int kb = 0; kb < D_; kb += 32) {
    __syncthreads();
    for (int i = tid; i < 256; i += 128) {       // A: 64 rows x 4 uint4
      int r = i >> 2, c = i & 3;
      *(uint4*)&Al[r * 40 + c * 8] =
          *(const uint4*)(oh + (size_t)(row0 + r) * D_ + kb + c * 8);
    }
    for (int i = tid; i < 2048; i += 128) {      // B: 64 x 32 f32->f16
      int n = i >> 5, kk = i & 31;
      Bl[n * 40 + kk] = (h16)Wo[(size_t)(n0 + n) * D_ + kb + kk];
    }
    if (kb + 32 < D_)   // pull next Wo tile toward L2 (global_prefetch_b8)
      __builtin_prefetch(Wo + (size_t)(n0 + (tid >> 1)) * D_ + kb + 32 + (tid & 1) * 16, 0, 0);
    __syncthreads();

    v16h a = ld_a(Al, 40, wave * 16 + (lane & 15), 0);
#pragma unroll
    for (int nt = 0; nt < 4; ++nt) {
      v16h bb = ld_b_nk(Bl, 40, nt * 16 + (lane & 15), 0);
      acc[nt] = __builtin_amdgcn_wmma_f32_16x16x32_f16(false, a, false, bb,
                                                       (short)0, acc[nt], false, false);
    }
  }

#pragma unroll
  for (int nt = 0; nt < 4; ++nt) {
    int col = n0 + nt * 16 + (lane & 15);
    float bb = bo[col];
#pragma unroll
    for (int r = 0; r < 8; ++r) {
      int row = row0 + wave * 16 + r + ((lane < 16) ? 0 : 8);
      out[(size_t)row * D_ + col] = acc[nt][r] + bb;
    }
  }
}

// ---------------- host launcher ----------------
extern "C" void kernel_launch(void* const* d_in, const int* in_sizes, int n_in,
                              void* d_out, int out_size, void* d_ws, size_t ws_size,
                              hipStream_t stream) {
  (void)in_sizes; (void)n_in; (void)out_size; (void)ws_size;
  const float* q  = (const float*)d_in[0];
  const float* k  = (const float*)d_in[1];
  const float* v  = (const float*)d_in[2];
  const float* Wq = (const float*)d_in[3];
  const float* bq = (const float*)d_in[4];
  const float* Wk = (const float*)d_in[5];
  const float* bk = (const float*)d_in[6];
  const float* Wv = (const float*)d_in[7];
  const float* bv = (const float*)d_in[8];
  const float* Wo = (const float*)d_in[9];
  const float* bo = (const float*)d_in[10];

  const size_t NPH = (size_t)B_ * H_ * S_ * DK_;
  h16* qp = (h16*)d_ws;
  h16* kp = qp + NPH;
  h16* vp = kp + NPH;
  h16* oh = vp + NPH;          // [B,S,D] f16, heads concatenated

  proj_kernel<<<dim3(S_ / 64, B_ * H_, 3), 128, 0, stream>>>(
      q, k, v, Wq, bq, Wk, bk, Wv, bv, qp, kp, vp);
  attn_kernel<<<dim3(S_ / 64, B_ * H_), 128, 0, stream>>>(qp, kp, vp, oh);
  outproj_kernel<<<dim3((B_ * S_) / 64, D_ / 64), 128, 0, stream>>>(
      oh, Wo, bo, (float*)d_out);
}
